// MKR_71468255805675
// MI455X (gfx1250) — matrix operand order repaired and checked by reading, not compiled, not run on hardware
//
#include <hip/hip_runtime.h>
#include <hip/hip_bf16.h>

#define D 128
#define NLAYER 3
#define WAVES 8
#define ROWS_PER_BLOCK 128   // WAVES * 16

typedef __attribute__((ext_vector_type(4)))  __bf16 v4bf;
typedef __attribute__((ext_vector_type(8)))  __bf16 v8bf;
typedef __attribute__((ext_vector_type(16))) __bf16 v16bf;
typedef __attribute__((ext_vector_type(8)))  float  v8f;
typedef __attribute__((ext_vector_type(4)))  int    v4i;

#define AS1 __attribute__((address_space(1)))
#define AS3 __attribute__((address_space(3)))

// ---- d_ws layout (written once by prep kernel, byte offsets) ----
//   [0      .. 98304)  : W bf16, [3][128][128]  W[layer][n][k]
//   [98304  .. 105984) : consts f32[1920] = wvsum[3][128] | wesum[3][128]
//                        | bv[3][128] | be[3][128] | mlpb[3][128]
#define WS_W_BYTES     (NLAYER * D * D * 2)          // 98304
#define WS_C_BYTES     (5 * NLAYER * D * 4)          // 7680
#define WS_STAGE_BYTES (WS_W_BYTES + WS_C_BYTES)     // 105984

// ---- LDS layout: mirrors the staged ws region, then per-wave tiles ----
//   sW    : bytes [0      .. 98304)
//   sC    : bytes [98304  .. 105984)
//   sAct  : bytes [105984 .. 138752)   8*16*128 bf16
//   sV3   : bytes [138752 .. 171520)   8*16*128 bf16
#define SMEM_BYTES ((size_t)(WS_STAGE_BYTES + 2 * WAVES * 16 * D * 2))

// =====================================================================
// Prep kernel: convert mlp_W -> bf16 and fold CAC constants, once.
// grid 48 x 256 = 12288 threads (one float4 of W each).
// =====================================================================
__global__ __launch_bounds__(256) void mkr_prep_kernel(
    const float* __restrict__ mlp_W,
    const float* __restrict__ mlp_b,
    const float* __restrict__ w_vv,
    const float* __restrict__ w_ev,
    const float* __restrict__ w_ve,
    const float* __restrict__ b_v,
    const float* __restrict__ b_e,
    char*        __restrict__ ws)
{
    const int i = blockIdx.x * 256 + threadIdx.x;
    __bf16* wbf = (__bf16*)ws;
    float*  c   = (float*)(ws + WS_W_BYTES);

    if (i < (NLAYER * D * D) / 4) {
        float4 w = ((const float4*)mlp_W)[i];
        v4bf bb;
        bb[0] = (__bf16)w.x; bb[1] = (__bf16)w.y;
        bb[2] = (__bf16)w.z; bb[3] = (__bf16)w.w;
        *(v4bf*)(wbf + i * 4) = bb;
    }
    if (i < NLAYER * D) {
        c[i]        = w_vv[i] + w_ev[i];   // wvsum
        c[384 + i]  = w_ve[i] + w_vv[i];   // wesum
        c[768 + i]  = b_v[i];
        c[1152 + i] = b_e[i];
        c[1536 + i] = mlp_b[i];
    }
}

// =====================================================================
// Main fused kernel.
// =====================================================================
__global__ __launch_bounds__(256) void mkr_fused_kernel(
    const int*   __restrict__ user_idx,
    const int*   __restrict__ item_idx,
    const float* __restrict__ user_emb,
    const float* __restrict__ item_emb,
    const float* __restrict__ entity_emb,
    const char*  __restrict__ ws,
    float*       __restrict__ out)
{
    extern __shared__ char smem[];
    __bf16* sW   = (__bf16*)smem;                     // [3][128][128] bf16
    float*  sC   = (float*)(smem + WS_W_BYTES);       // f32[1920]
    __bf16* sAct = (__bf16*)(smem + WS_STAGE_BYTES);  // [8][16][128]
    __bf16* sV3  = sAct + WAVES * 16 * D;             // [8][16][128]
    float* sWV = sC;            // wvsum [3][128]
    float* sWE = sC + 384;      // wesum
    float* sBV = sC + 768;      // b_v
    float* sBE = sC + 1152;     // b_e
    float* sMB = sC + 1536;     // mlp_b

    const int tid = threadIdx.x;

    // -------- Phase 0: stage prepacked W+consts into LDS (async path) -----
    {
        const int nchunk = WS_STAGE_BYTES / 16;   // 6624 x 16B
#if __has_builtin(__builtin_amdgcn_global_load_async_to_lds_b128)
        for (int i = tid; i < nchunk; i += 256) {
            __builtin_amdgcn_global_load_async_to_lds_b128(
                (AS1 v4i*)(ws + (size_t)i * 16),
                (AS3 v4i*)(smem + (size_t)i * 16),
                /*offset=*/0, /*cpol=*/0);
        }
#if __has_builtin(__builtin_amdgcn_s_wait_asynccnt)
        __builtin_amdgcn_s_wait_asynccnt(0);
#else
        asm volatile("s_wait_asynccnt 0x0" ::: "memory");
#endif
#else
        const uint4* src = (const uint4*)ws;
        for (int i = tid; i < nchunk; i += 256)
            *(uint4*)(smem + (size_t)i * 16) = src[i];
#endif
    }
    __syncthreads();

    const int lane = tid & 31;
    const int wave = tid >> 5;
    const int row  = lane & 15;   // batch row within the wave's 16-row tile
    const int hlf  = lane >> 4;   // which 64-element half of the 128-dim
    const int col0 = row;         // (lane & 15): column residue in fragments

    const int bidx = blockIdx.x * ROWS_PER_BLOCK + wave * 16 + row;
    const int ui = user_idx[bidx];
    const int ii = item_idx[bidx];

    __bf16* actBase = sAct + wave * (16 * D);
    __bf16* v3Base  = sV3  + wave * (16 * D);

    // -------- Phase 1: gather u -> LDS(bf16); CAC3 via scalar identity ----
    {
        const float4* up = (const float4*)(user_emb   + (size_t)ui * D + hlf * 64);
        const float4* vp = (const float4*)(item_emb   + (size_t)ii * D + hlf * 64);
        const float4* ep = (const float4*)(entity_emb + (size_t)ii * D + hlf * 64);
        const float4* wv0 = (const float4*)(sWV + hlf * 64);
        const float4* we0 = (const float4*)(sWE + hlf * 64);

        __bf16* actRow = actBase + row * D + hlf * 64;

        float4 ee[16];
        float pv = 0.f, pe = 0.f;
#pragma unroll
        for (int j = 0; j < 16; ++j) {
            float4 u = up[j];
            v4bf ub;
            ub[0] = (__bf16)u.x; ub[1] = (__bf16)u.y;
            ub[2] = (__bf16)u.z; ub[3] = (__bf16)u.w;
            *(v4bf*)(actRow + j * 4) = ub;

            float4 v = vp[j];
            ee[j] = ep[j];
            float4 a = wv0[j], c = we0[j];
            pv += v.x * a.x + v.y * a.y + v.z * a.z + v.w * a.w;
            pe += v.x * c.x + v.y * c.y + v.z * c.z + v.w * c.w;
        }
        float sv = pv + __shfl_xor(pv, 16, 32);   // s_v0
        float se = pe + __shfl_xor(pe, 16, 32);   // s_e0

        // v_i = e_{i-1}*sv_{i-1} + bv_{i-1}  => dots of v_i reduce to dots of e
#pragma unroll
        for (int L = 1; L < NLAYER; ++L) {
            const float4* wvL = (const float4*)(sWV + L * D + hlf * 64);
            const float4* weL = (const float4*)(sWE + L * D + hlf * 64);
            const float4* bvP = (const float4*)(sBV + (L - 1) * D + hlf * 64);
            const float4* beP = (const float4*)(sBE + (L - 1) * D + hlf * 64);
            float d1 = 0.f, d2 = 0.f, cv = 0.f, ce = 0.f;
#pragma unroll
            for (int j = 0; j < 16; ++j) {
                float4 e = ee[j];
                float4 a = wvL[j], c = weL[j], bv = bvP[j];
                d1 += e.x * a.x + e.y * a.y + e.z * a.z + e.w * a.w;
                d2 += e.x * c.x + e.y * c.y + e.z * c.z + e.w * c.w;
                cv += bv.x * a.x + bv.y * a.y + bv.z * a.z + bv.w * a.w;
                ce += bv.x * c.x + bv.y * c.y + bv.z * c.z + bv.w * c.w;
            }
            d1 += __shfl_xor(d1, 16, 32);
            d2 += __shfl_xor(d2, 16, 32);
            cv += __shfl_xor(cv, 16, 32);
            ce += __shfl_xor(ce, 16, 32);
            float svN = sv * d1 + cv;   // s_v[L]
            float seN = sv * d2 + ce;   // s_e[L]
            // e_L = e_{L-1} * s_e[L-1] + b_e[L-1]
#pragma unroll
            for (int j = 0; j < 16; ++j) {
                float4 e = ee[j], be = beP[j];
                ee[j] = make_float4(e.x * se + be.x, e.y * se + be.y,
                                    e.z * se + be.z, e.w * se + be.w);
            }
            sv = svN;
            se = seN;
        }
        // v3 = e2 * s_v2 + b_v2 -> bf16 LDS
        const float4* bv2 = (const float4*)(sBV + 2 * D + hlf * 64);
        __bf16* v3Row = v3Base + row * D + hlf * 64;
#pragma unroll
        for (int j = 0; j < 16; ++j) {
            float4 e = ee[j], bv = bv2[j];
            v4bf o;
            o[0] = (__bf16)(e.x * sv + bv.x);
            o[1] = (__bf16)(e.y * sv + bv.y);
            o[2] = (__bf16)(e.z * sv + bv.z);
            o[3] = (__bf16)(e.w * sv + bv.w);
            *(v4bf*)(v3Row + j * 4) = o;
        }
    }

    // -------- Phase 2: 3-layer MLP via bf16 WMMA on per-wave 16x128 tile --
    float outPart[8];
#pragma unroll
    for (int layer = 0; layer < NLAYER; ++layer) {
        v8f acc[8];
#pragma unroll
        for (int t = 0; t < 8; ++t) {
            float bz = sMB[layer * D + t * 16 + col0];   // bias, col-indexed
            acc[t] = (v8f){bz, bz, bz, bz, bz, bz, bz, bz};
        }
#pragma unroll
        for (int kk = 0; kk < 4; ++kk) {
            const int k0 = kk * 32;
            // A fragment: lane holds row (lane&15); 16 contiguous K (k0 + hlf*16)
            const __bf16* ar = actBase + row * D + k0 + hlf * 16;
            v8bf c1 = *(const v8bf*)(ar);        // K = k0 + hlf*16 .. +8
            v8bf c2 = *(const v8bf*)(ar + 8);    // K = k0+hlf*16+8 .. +8
            v16bf aF = __builtin_shufflevector(c1, c2,
                0, 1, 2, 3, 4, 5, 6, 7, 8, 9, 10, 11, 12, 13, 14, 15);
#pragma unroll
            for (int t = 0; t < 8; ++t) {
                // B fragment: lane n -> column (n&15), 16 contiguous K of W row
                const __bf16* br = sW + layer * (D * D)
                                 + (t * 16 + col0) * D + k0 + hlf * 16;
                v16bf bF = *(const v16bf*)br;
                acc[t] = __builtin_amdgcn_wmma_f32_16x16x32_bf16(
                    false, aF, false, bF, (short)0, acc[t], false, false);
            }
        }
        if (layer < NLAYER - 1) {
            // ReLU + write back activations (bf16) for next layer's A frags
#pragma unroll
            for (int t = 0; t < 8; ++t) {
#pragma unroll
                for (int r = 0; r < 8; ++r) {
                    float x = acc[t][r];
                    x = x > 0.f ? x : 0.f;
                    actBase[(r + hlf * 8) * D + t * 16 + col0] = (__bf16)x;
                }
            }
        } else {
            // Final: out[b] = dot(relu(u3), v3) straight from C-frag layout
#pragma unroll
            for (int r = 0; r < 8; ++r) outPart[r] = 0.f;
#pragma unroll
            for (int t = 0; t < 8; ++t) {
#pragma unroll
                for (int r = 0; r < 8; ++r) {
                    float x = acc[t][r];
                    x = x > 0.f ? x : 0.f;
                    float v3 = (float)v3Base[(r + hlf * 8) * D + t * 16 + col0];
                    outPart[r] += x * v3;
                }
            }
#pragma unroll
            for (int r = 0; r < 8; ++r) {
                float red = outPart[r];
                red += __shfl_xor(red, 1, 32);
                red += __shfl_xor(red, 2, 32);
                red += __shfl_xor(red, 4, 32);
                red += __shfl_xor(red, 8, 32);
                if (col0 == 0)
                    out[blockIdx.x * ROWS_PER_BLOCK + wave * 16 + hlf * 8 + r] = red;
            }
        }
    }
}

extern "C" void kernel_launch(void* const* d_in, const int* in_sizes, int n_in,
                              void* d_out, int out_size, void* d_ws, size_t ws_size,
                              hipStream_t stream) {
    const int*   user_idx   = (const int*)d_in[0];
    const int*   item_idx   = (const int*)d_in[1];
    const float* user_emb   = (const float*)d_in[2];
    const float* item_emb   = (const float*)d_in[3];
    const float* entity_emb = (const float*)d_in[4];
    const float* mlp_W      = (const float*)d_in[5];
    const float* mlp_b      = (const float*)d_in[6];
    const float* w_vv       = (const float*)d_in[7];
    const float* w_ev       = (const float*)d_in[8];
    const float* w_ve       = (const float*)d_in[9];
    const float* b_v        = (const float*)d_in[10];
    const float* b_e        = (const float*)d_in[11];
    float* out = (float*)d_out;
    char*  ws  = (char*)d_ws;

    const int B = in_sizes[0];                 // 131072
    const int grid = B / ROWS_PER_BLOCK;       // 1024

    (void)hipFuncSetAttribute((const void*)mkr_fused_kernel,
                              hipFuncAttributeMaxDynamicSharedMemorySize,
                              (int)SMEM_BYTES);

    mkr_prep_kernel<<<48, 256, 0, stream>>>(
        mlp_W, mlp_b, w_vv, w_ev, w_ve, b_v, b_e, ws);

    mkr_fused_kernel<<<grid, 256, SMEM_BYTES, stream>>>(
        user_idx, item_idx, user_emb, item_emb, entity_emb, ws, out);
}
